// LocalConceptTransforming_8126078124234
// MI455X (gfx1250) — matrix operand order repaired and checked by reading, not compiled, not run on hardware
//
#include <hip/hip_runtime.h>
#include <hip/hip_bf16.h>
#include <math.h>

// ---------------------------------------------------------------------------
// Problem constants (from the reference)
// ---------------------------------------------------------------------------
constexpr int cL      = 77;
constexpr int cN      = 2048;
constexpr int cD      = 512;
constexpr int cC      = 64;
constexpr int cTOPK   = 8;
constexpr int cLAYERS = 2;
constexpr int cHEADS  = 8;
constexpr int cT      = cC * cN;          // 131072 tokens in the transformer

typedef __attribute__((ext_vector_type(16))) __bf16 v16bf;
typedef __attribute__((ext_vector_type(8)))  float  v8f;

static __device__ __forceinline__ v8f wmma_bf16(v16bf a, v16bf b, v8f c) {
  // D = A(16x32 bf16) * B(32x16 bf16) + C(16x16 f32)
  return __builtin_amdgcn_wmma_f32_16x16x32_bf16(false, a, false, b,
                                                 (short)0, c, false, false);
}

static __device__ __forceinline__ v8f zero8() {
  v8f z = {0.f, 0.f, 0.f, 0.f, 0.f, 0.f, 0.f, 0.f};
  return z;
}

// Build a fragment from two contiguous 8 x bf16 (16-byte) runs.
// Per ISA 7.12.2, a lane's A-fragment elements 0..7 are K kb0..kb0+7 and
// elements 8..15 are K kb1..kb1+7 (kb0/kb1 depend on lane half) — i.e. two
// 128-bit vector loads. Callers here use kb0 = 16*half, kb1 = kb0+8 for both A and B.
static __device__ __forceinline__ v16bf frag_2x8(const __bf16* p0,
                                                 const __bf16* p1) {
  v16bf a;
  *(uint4*)&a       = *(const uint4*)p0;
  *((uint4*)&a + 1) = *(const uint4*)p1;
  return a;
}

// ---------------------------------------------------------------------------
// Tensor Data Mover: 2D tile (rows x rowlen bf16, row stride in elements)
// from global memory into dense row-major LDS.  D# layout per
// CDNA5 ISA.  6-arg builtin form (clang-23):
//   (uint32x4 g0, int32x8 g1, int32x4 g2, int32x4 g3, int32x8 pad, i32 cpol)
// ---------------------------------------------------------------------------
#if defined(__gfx1250__) && __has_builtin(__builtin_amdgcn_tensor_load_to_lds)
#define USE_TDM 1
typedef __attribute__((ext_vector_type(4))) unsigned int u32x4;
typedef __attribute__((ext_vector_type(8))) int i32x8;
typedef __attribute__((ext_vector_type(4))) int i32x4;

static __device__ __forceinline__ void tdm_load_2d_bf16(
    uint32_t lds_off, const __bf16* gptr, uint32_t rows, uint32_t rowlen,
    uint32_t row_stride_elems) {
  uint64_t ga = (uint64_t)(uintptr_t)gptr;
  u32x4 g0;
  g0[0] = 1u;                                     // count=1 (valid), user mode
  g0[1] = lds_off;                                // lds_addr (bytes)
  g0[2] = (uint32_t)(ga & 0xffffffffu);           // global_addr[31:0]
  g0[3] = (uint32_t)((ga >> 32) & 0x01ffffffu)    // global_addr[56:32]
        | (2u << 30);                             // type = 2 ("image")
  i32x8 g1;
  g1[0] = (int)(1u << 16);                        // data_size=1 -> 2 bytes
  g1[1] = (int)(rowlen << 16);                    // tensor_dim0[15:0]
  g1[2] = (int)((rowlen >> 16) | (rows << 16));   // dim0 hi | tensor_dim1 lo
  g1[3] = (int)((rows >> 16) | (rowlen << 16));   // dim1 hi | tile_dim0
  g1[4] = (int)rows;                              // tile_dim1 (tile_dim2=0)
  g1[5] = (int)row_stride_elems;                  // tensor_dim0_stride[31:0]
  g1[6] = 0;
  g1[7] = 0;
  i32x4 gz4 = {0, 0, 0, 0};
  i32x8 gz8 = {0, 0, 0, 0, 0, 0, 0, 0};
  __builtin_amdgcn_tensor_load_to_lds(g0, g1, gz4, gz4, gz8, 0);
}
#endif

// ---------------------------------------------------------------------------
// 0) Weight f32 -> bf16 pre-conversion (weights are L2-resident afterwards)
// ---------------------------------------------------------------------------
__global__ void wcvt_kernel(const float* __restrict__ src,
                            __bf16* __restrict__ dst, int n4) {
  int i = blockIdx.x * blockDim.x + threadIdx.x;
  if (i >= n4) return;
  float4 v = ((const float4*)src)[i];
  __bf16* o = dst + (size_t)i * 4;
  o[0] = (__bf16)v.x; o[1] = (__bf16)v.y; o[2] = (__bf16)v.z; o[3] = (__bf16)v.w;
}

// ---------------------------------------------------------------------------
// 1) LTA: top-k threshold + pseudo label
// ---------------------------------------------------------------------------
__global__ void lta_mask_kernel(const float* __restrict__ tce,
                                float* __restrict__ attbuf,
                                float* __restrict__ pl) {
  const float NEG = -__builtin_inff();
  int ln = blockIdx.x;            // (l, n) pair, 0 .. L*N-1
  int c  = threadIdx.x;           // 0..63
  __shared__ float work[cC];
  __shared__ float red[cC];

  float s = tce[(size_t)ln * cC + c];
  float v = (s > 0.f) ? s : NEG;
  work[c] = v;
  __syncthreads();

  float thr = NEG;
  for (int it = 0; it < cTOPK; ++it) {
    red[c] = work[c];
    __syncthreads();
    for (int off = 32; off > 0; off >>= 1) {
      if (c < off) red[c] = fmaxf(red[c], red[c + off]);
      __syncthreads();
    }
    float m = red[0];
    if (it == cTOPK - 1) {
      thr = m;
    } else if (c == 0) {
      for (int j = 0; j < cC; ++j)
        if (work[j] == m) { work[j] = NEG; break; }
    }
    __syncthreads();
  }

  float masked = (v >= thr) ? v : NEG;
  attbuf[(size_t)ln * cC + c] = masked;
  pl[(size_t)ln * cC + c]     = (masked > 0.f) ? 1.f : 0.f;
}

// Softmax over the L axis, in place; all -inf column -> zeros (NaN guard).
__global__ void lta_softmax_kernel(float* __restrict__ attbuf) {
  const float NEG = -__builtin_inff();
  int id = blockIdx.x * blockDim.x + threadIdx.x;   // n*C + c
  if (id >= cN * cC) return;
  const size_t stride = (size_t)cN * cC;
  float mx = NEG;
  for (int l = 0; l < cL; ++l) mx = fmaxf(mx, attbuf[l * stride + id]);
  if (mx == NEG) {
    for (int l = 0; l < cL; ++l) attbuf[l * stride + id] = 0.f;
    return;
  }
  float sum = 0.f;
  for (int l = 0; l < cL; ++l) sum += expf(attbuf[l * stride + id] - mx);
  float inv = 1.f / sum;
  for (int l = 0; l < cL; ++l)
    attbuf[l * stride + id] = expf(attbuf[l * stride + id] - mx) * inv;
}

// ---------------------------------------------------------------------------
// 2) LTA merge GEMM: h[c,n,d] = sum_l att[l,n,c] * x[l,n,d]  (+ positional enc)
//    Per batch n: [C=64, K=77] x [77, D=512], WMMA bf16, K padded/guarded.
// ---------------------------------------------------------------------------
static __device__ __forceinline__ float pe_val(int c, int d) {
  float arg = (float)c * expf(-(float)(d & ~1) * (logf(10000.f) / (float)cD));
  float v = (d & 1) ? cosf(arg) : sinf(arg);
  return v * 0.04419417382415922f;   // 1/sqrt(512)
}

__global__ void lta_merge_kernel(const float* __restrict__ attbuf,
                                 const float* __restrict__ x,
                                 float* __restrict__ h) {
  int nb = blockIdx.y;
  int d0 = blockIdx.x * 32;
  int w = threadIdx.x >> 5, lane = threadIdx.x & 31;
  int mi = w & 3, ni = w >> 2;
  int cbase = mi * 16;
  int dbase = d0 + ni * 16;
  int m = lane & 15, hi = lane >> 4;

  const size_t strideA = (size_t)cN * cC;   // att l-stride
  const size_t strideB = (size_t)cN * cD;   // x   l-stride
  const float* ap = attbuf + (size_t)nb * cC + (cbase + m);
  const float* bp = x + (size_t)nb * cD + (dbase + (lane & 15));

  v8f acc = zero8();
  for (int kc = 0; kc < 96; kc += 32) {     // K=77 padded to 96
    v16bf a;
#pragma unroll
    for (int r = 0; r < 8; ++r) {
      int kb = 2 * r + hi * 16;
      int l0 = kc + kb, l1 = l0 + 1;
      a[2 * r]     = (l0 < cL) ? (__bf16)ap[(size_t)l0 * strideA] : (__bf16)0.f;
      a[2 * r + 1] = (l1 < cL) ? (__bf16)ap[(size_t)l1 * strideA] : (__bf16)0.f;
    }
    v16bf b;
#pragma unroll
    for (int e = 0; e < 16; ++e) {
      int l = kc + hi * 16 + e;
      b[e] = (l < cL) ? (__bf16)bp[(size_t)l * strideB] : (__bf16)0.f;
    }
    acc = wmma_bf16(a, b, acc);
  }

  int dcol = dbase + (lane & 15);
#pragma unroll
  for (int r = 0; r < 8; ++r) {
    int c = cbase + hi * 8 + r;
    h[((size_t)c * cN + nb) * cD + dcol] = acc[r] + pe_val(c, dcol);
  }
}

// ---------------------------------------------------------------------------
// 3) LDS-staged, double-buffered bf16 WMMA GEMM, K-step 64 (8 WMMA / barrier)
//    Out = epilogue(A[T,K] * Wb^T + bias);  Wb bf16 [Ncols, K] (K-contiguous).
//    Block tile 128x64, 8 waves, wave tile 32x32 (4 WMMA accumulators).
//    LDS rows padded to 72 elems (144B) for bank-conflict-free b128 reads.
//    mode 0: store bf16; mode 1: quickGELU -> bf16; mode 2: h += (f32 residual)
// ---------------------------------------------------------------------------
__global__ void gemm_bf16_kernel(const __bf16* __restrict__ A, int K,
                                 const __bf16* __restrict__ Wb,
                                 const float* __restrict__ bias,
                                 void* __restrict__ outp, int ldout, int mode) {
  __shared__ alignas(16) __bf16 As[2][128][72];
  __shared__ alignas(16) __bf16 Bs[2][64][72];

  const int tid  = threadIdx.x;
  const int row0 = blockIdx.y * 128;
  const int col0 = blockIdx.x * 64;
  const int w    = tid >> 5;
  const int lane = tid & 31;
  const int wm   = w & 3;            // 4 m-waves  -> 32 rows each
  const int wn   = w >> 2;           // 2 n-waves  -> 32 cols each
  const int nlane = lane & 15;
  const int hi    = lane >> 4;

  v8f acc[2][2] = {{zero8(), zero8()}, {zero8(), zero8()}};

  auto stage = [&](int s, int kc) {
    // A tile: 128 rows x 64 k, uint4 (8 bf16) per load, 4 per thread
#pragma unroll
    for (int j = 0; j < 4; ++j) {
      int idx = tid + j * 256;                 // 0..1023
      int row = idx >> 3, seg = idx & 7;
      *(uint4*)&As[s][row][seg * 8] =
          *(const uint4*)(A + (size_t)(row0 + row) * K + kc + seg * 8);
    }
    // B tile: 64 cols x 64 k (K-contiguous weight rows), 2 per thread
#pragma unroll
    for (int j = 0; j < 2; ++j) {
      int idx = tid + j * 256;                 // 0..511
      int n = idx >> 3, seg = idx & 7;
      *(uint4*)&Bs[s][n][seg * 8] =
          *(const uint4*)(Wb + (size_t)(col0 + n) * K + kc + seg * 8);
    }
  };

  stage(0, 0);
  __syncthreads();

  int s = 0;
  for (int kc = 0; kc < K; kc += 64) {
    if (kc + 64 < K) {
      __builtin_prefetch((const void*)(A + (size_t)(row0 + (tid >> 3)) * K + kc + 128), 0, 1);
      stage(s ^ 1, kc + 64);
    }
#pragma unroll
    for (int c = 0; c < 2; ++c) {            // two 32-K chunks per stage
      v16bf af[2], bfr[2];
#pragma unroll
      for (int mt = 0; mt < 2; ++mt) {
        const __bf16* rp = &As[s][wm * 32 + mt * 16 + nlane][c * 32];
        af[mt] = frag_2x8(rp + hi * 16, rp + hi * 16 + 8);
      }
#pragma unroll
      for (int nt = 0; nt < 2; ++nt) {
        const __bf16* cp = &Bs[s][wn * 32 + nt * 16 + nlane][c * 32 + hi * 16];
        bfr[nt] = frag_2x8(cp, cp + 8);
      }
#pragma unroll
      for (int mt = 0; mt < 2; ++mt)
#pragma unroll
        for (int nt = 0; nt < 2; ++nt)
          acc[mt][nt] = wmma_bf16(af[mt], bfr[nt], acc[mt][nt]);
    }
    __syncthreads();
    s ^= 1;
  }

#pragma unroll
  for (int mt = 0; mt < 2; ++mt) {
#pragma unroll
    for (int nt = 0; nt < 2; ++nt) {
      int rowb = row0 + wm * 32 + mt * 16 + hi * 8;
      int col  = col0 + wn * 32 + nt * 16 + nlane;
      float bv = bias[col];
      if (mode == 2) {
        float* o = (float*)outp;
#pragma unroll
        for (int r = 0; r < 8; ++r) {
          size_t idx = (size_t)(rowb + r) * ldout + col;
          o[idx] = o[idx] + acc[mt][nt][r] + bv;   // residual into h
        }
      } else {
        __bf16* o = (__bf16*)outp;
#pragma unroll
        for (int r = 0; r < 8; ++r) {
          float v = acc[mt][nt][r] + bv;
          if (mode == 1) v = v / (1.f + expf(-1.702f * v));  // quickGELU
          o[(size_t)(rowb + r) * ldout + col] = (__bf16)v;
        }
      }
    }
  }
}

// ---------------------------------------------------------------------------
// 4) Fused attention per (n, head): S=64, hd=64.  QKV tiles staged into LDS
//    by the Tensor Data Mover (fallback: vector copies), then
//    scores = Q*K^T/8 (WMMA) -> softmax -> P*V (WMMA).
// ---------------------------------------------------------------------------
__global__ void attn_kernel(const __bf16* __restrict__ qkv,
                            __bf16* __restrict__ aout) {
  int head = blockIdx.x;
  int nb   = blockIdx.y;
  __shared__ alignas(16) __bf16 Qs[64 * 64];
  __shared__ alignas(16) __bf16 Ks[64 * 64];
  __shared__ alignas(16) __bf16 Vs[64 * 64];
  __shared__ alignas(16) __bf16 Ps[64 * 64];
  __shared__ float Ss[64 * 64];

  int tid = threadIdx.x;
  {
    size_t gbase = (size_t)nb * (3 * cD) + head * 64;   // token (s=0, nb)
#ifdef USE_TDM
    if (tid < 32) {   // one wave issues the three tile DMAs
      tdm_load_2d_bf16((uint32_t)(uintptr_t)&Qs[0], qkv + gbase,          64, 64, 3 * cD);
      tdm_load_2d_bf16((uint32_t)(uintptr_t)&Ks[0], qkv + gbase + cD,     64, 64, 3 * cD);
      tdm_load_2d_bf16((uint32_t)(uintptr_t)&Vs[0], qkv + gbase + 2 * cD, 64, 64, 3 * cD);
      __builtin_amdgcn_s_wait_tensorcnt(0);
    }
#else
    for (int idx = tid; idx < 512; idx += 256) {        // 8 bf16 per copy
      int sq = idx >> 3, seg = (idx & 7) * 8;
      size_t g = gbase + (size_t)sq * cN * (3 * cD) + seg;
      *(uint4*)&Qs[sq * 64 + seg] = *(const uint4*)&qkv[g];
      *(uint4*)&Ks[sq * 64 + seg] = *(const uint4*)&qkv[g + cD];
      *(uint4*)&Vs[sq * 64 + seg] = *(const uint4*)&qkv[g + 2 * cD];
    }
#endif
  }
  __syncthreads();

  int w = tid >> 5, lane = tid & 31, nlane = lane & 15, hi = lane >> 4;

  // scores = Q * K^T * 0.125 ; B[k=d][n=t'] = Ks[t'][d] is K-contiguous
  for (int t = w; t < 16; t += 8) {
    int mi = t >> 2, ni = t & 3;
    v8f acc = zero8();
#pragma unroll
    for (int kc = 0; kc < 64; kc += 32) {
      const __bf16* rp = Qs + (size_t)(mi * 16 + nlane) * 64 + kc;
      v16bf a = frag_2x8(rp + hi * 16, rp + hi * 16 + 8);
      const __bf16* cp = Ks + (size_t)(ni * 16 + nlane) * 64 + kc + hi * 16;
      v16bf b = frag_2x8(cp, cp + 8);
      acc = wmma_bf16(a, b, acc);
    }
#pragma unroll
    for (int r = 0; r < 8; ++r)
      Ss[(mi * 16 + hi * 8 + r) * 64 + ni * 16 + nlane] = acc[r] * 0.125f;
  }
  __syncthreads();

  if (tid < 64) {                      // row softmax
    float mx = Ss[tid * 64];
    for (int j = 1; j < 64; ++j) mx = fmaxf(mx, Ss[tid * 64 + j]);
    float sum = 0.f;
    for (int j = 0; j < 64; ++j) sum += expf(Ss[tid * 64 + j] - mx);
    float inv = 1.f / sum;
    for (int j = 0; j < 64; ++j)
      Ps[tid * 64 + j] = (__bf16)(expf(Ss[tid * 64 + j] - mx) * inv);
  }
  __syncthreads();

  // out = P * V ; B[k=t'][n=d] = Vs[t'][d] is k-strided (transposed operand)
  for (int t = w; t < 16; t += 8) {
    int mi = t >> 2, ni = t & 3;
    v8f acc = zero8();
#pragma unroll
    for (int kc = 0; kc < 64; kc += 32) {
      const __bf16* rp = Ps + (size_t)(mi * 16 + nlane) * 64 + kc;
      v16bf a = frag_2x8(rp + hi * 16, rp + hi * 16 + 8);
      v16bf b;
#pragma unroll
      for (int e = 0; e < 16; ++e)
        b[e] = Vs[(size_t)(kc + hi * 16 + e) * 64 + ni * 16 + nlane];
      acc = wmma_bf16(a, b, acc);
    }
    int col = ni * 16 + nlane;
#pragma unroll
    for (int r = 0; r < 8; ++r) {
      int row = mi * 16 + hi * 8 + r;    // query position s
      aout[((size_t)row * cN + nb) * cD + head * 64 + col] = (__bf16)acc[r];
    }
  }
}

// ---------------------------------------------------------------------------
// 5) LayerNorm: one wave32 per 512-wide row; b128 loads; bf16 output.
// ---------------------------------------------------------------------------
__global__ void layernorm_kernel(const float* __restrict__ h,
                                 const float* __restrict__ g,
                                 const float* __restrict__ b,
                                 __bf16* __restrict__ out) {
  int row  = blockIdx.x * 8 + (threadIdx.x >> 5);
  int lane = threadIdx.x & 31;
  const float* x = h + (size_t)row * cD + lane * 16;
  float buf[16];
  float s = 0.f, s2 = 0.f;
#pragma unroll
  for (int j = 0; j < 4; ++j) {
    float4 v = ((const float4*)x)[j];
    buf[4 * j]     = v.x; buf[4 * j + 1] = v.y;
    buf[4 * j + 2] = v.z; buf[4 * j + 3] = v.w;
    s  += v.x + v.y + v.z + v.w;
    s2 += v.x * v.x + v.y * v.y + v.z * v.z + v.w * v.w;
  }
#pragma unroll
  for (int off = 16; off > 0; off >>= 1) {
    s  += __shfl_xor(s, off);
    s2 += __shfl_xor(s2, off);
  }
  float mu  = s * (1.f / cD);
  float var = s2 * (1.f / cD) - mu * mu;
  float inv = rsqrtf(var + 1e-5f);
  __bf16* o = out + (size_t)row * cD + lane * 16;
#pragma unroll
  for (int j = 0; j < 16; ++j)
    o[j] = (__bf16)((buf[j] - mu) * inv * g[lane * 16 + j] + b[lane * 16 + j]);
}

// ---------------------------------------------------------------------------
// 6) Hash head: codes[n,k] = tanh(dot(h[k,n,:], hash_w[k,:]) + hash_b[k])
// ---------------------------------------------------------------------------
__global__ void hash_kernel(const float* __restrict__ h,
                            const float* __restrict__ hw,
                            const float* __restrict__ hb,
                            float* __restrict__ codes) {
  int id = blockIdx.x * blockDim.x + threadIdx.x;  // n*C + k
  if (id >= cN * cC) return;
  int k = id & (cC - 1);
  int n = id >> 6;
  const float4* hp = (const float4*)(h + ((size_t)k * cN + n) * cD);
  const float4* wp = (const float4*)(hw + (size_t)k * cD);
  float acc = 0.f;
  for (int d = 0; d < cD / 4; ++d) {
    float4 a = hp[d], bv = wp[d];
    acc += a.x * bv.x + a.y * bv.y + a.z * bv.z + a.w * bv.w;
  }
  codes[id] = tanhf(acc + hb[k]);
}

// ---------------------------------------------------------------------------
// Launch
// ---------------------------------------------------------------------------
extern "C" void kernel_launch(void* const* d_in, const int* in_sizes, int n_in,
                              void* d_out, int out_size, void* d_ws, size_t ws_size,
                              hipStream_t stream) {
  const float* x          = (const float*)d_in[0];
  const float* tce        = (const float*)d_in[1];
  const float* ln1_g      = (const float*)d_in[2];
  const float* ln1_b      = (const float*)d_in[3];
  const float* attn_in_w  = (const float*)d_in[4];
  const float* attn_in_b  = (const float*)d_in[5];
  const float* attn_out_w = (const float*)d_in[6];
  const float* attn_out_b = (const float*)d_in[7];
  const float* ln2_g      = (const float*)d_in[8];
  const float* ln2_b      = (const float*)d_in[9];
  const float* mlp_w1     = (const float*)d_in[10];
  const float* mlp_b1     = (const float*)d_in[11];
  const float* mlp_w2     = (const float*)d_in[12];
  const float* mlp_b2     = (const float*)d_in[13];
  const float* hash_w     = (const float*)d_in[14];
  const float* hash_b     = (const float*)d_in[15];

  float* codes = (float*)d_out;                        // [N, C]
  float* pl    = codes + (size_t)cN * cC;              // [L, N, C]
  float* h     = pl + (size_t)cL * cN * cC;            // [C, N, D] (live buffer)

  char* ws = (char*)d_ws;
  size_t off = 0;
  float*  attbuf  = (float*)(ws + off);  off += (size_t)cL * cN * cC * sizeof(float);
  __bf16* lnbuf   = (__bf16*)(ws + off); off += (size_t)cT * cD * sizeof(__bf16);
  __bf16* qkvbuf  = (__bf16*)(ws + off); off += (size_t)cT * (4 * cD) * sizeof(__bf16);
  __bf16* aoutbuf = (__bf16*)(ws + off); off += (size_t)cT * cD * sizeof(__bf16);
  // bf16 weights (all layers)
  __bf16* w_qkv = (__bf16*)(ws + off); off += (size_t)cLAYERS * 3 * cD * cD * sizeof(__bf16);
  __bf16* w_out = (__bf16*)(ws + off); off += (size_t)cLAYERS * cD * cD * sizeof(__bf16);
  __bf16* w_m1  = (__bf16*)(ws + off); off += (size_t)cLAYERS * 4 * cD * cD * sizeof(__bf16);
  __bf16* w_m2  = (__bf16*)(ws + off); off += (size_t)cLAYERS * cD * 4 * cD * sizeof(__bf16);

  auto cvt = [&](const float* src, __bf16* dst, size_t n) {
    int n4 = (int)(n / 4);
    wcvt_kernel<<<(n4 + 255) / 256, 256, 0, stream>>>(src, dst, n4);
  };
  cvt(attn_in_w,  w_qkv, (size_t)cLAYERS * 3 * cD * cD);
  cvt(attn_out_w, w_out, (size_t)cLAYERS * cD * cD);
  cvt(mlp_w1,     w_m1,  (size_t)cLAYERS * 4 * cD * cD);
  cvt(mlp_w2,     w_m2,  (size_t)cLAYERS * cD * 4 * cD);

  // --- LTA ---
  lta_mask_kernel<<<cL * cN, cC, 0, stream>>>(tce, attbuf, pl);
  lta_softmax_kernel<<<(cN * cC + 255) / 256, 256, 0, stream>>>(attbuf);
  lta_merge_kernel<<<dim3(cD / 32, cN), 256, 0, stream>>>(attbuf, x, h);

  // --- Transformer ---
  for (int i = 0; i < cLAYERS; ++i) {
    layernorm_kernel<<<cT / 8, 256, 0, stream>>>(h, ln1_g + i * cD, ln1_b + i * cD, lnbuf);
    gemm_bf16_kernel<<<dim3((3 * cD) / 64, cT / 128), 256, 0, stream>>>(
        lnbuf, cD, w_qkv + (size_t)i * 3 * cD * cD, attn_in_b + (size_t)i * 3 * cD,
        qkvbuf, 3 * cD, 0);
    attn_kernel<<<dim3(cHEADS, cN), 256, 0, stream>>>(qkvbuf, aoutbuf);
    gemm_bf16_kernel<<<dim3(cD / 64, cT / 128), 256, 0, stream>>>(
        aoutbuf, cD, w_out + (size_t)i * cD * cD, attn_out_b + (size_t)i * cD,
        h, cD, 2);
    layernorm_kernel<<<cT / 8, 256, 0, stream>>>(h, ln2_g + i * cD, ln2_b + i * cD, lnbuf);
    gemm_bf16_kernel<<<dim3((4 * cD) / 64, cT / 128), 256, 0, stream>>>(
        lnbuf, cD, w_m1 + (size_t)i * 4 * cD * cD, mlp_b1 + (size_t)i * 4 * cD,
        qkvbuf, 4 * cD, 1);
    gemm_bf16_kernel<<<dim3(cD / 64, cT / 128), 256, 0, stream>>>(
        qkvbuf, 4 * cD, w_m2 + (size_t)i * cD * 4 * cD, mlp_b2 + (size_t)i * cD,
        h, cD, 2);
  }

  // --- Hash head ---
  hash_kernel<<<(cN * cC + 255) / 256, 256, 0, stream>>>(h, hash_w, hash_b, codes);
}
